// SparseAttentionModel_25314537242835
// MI455X (gfx1250) — matrix operand order repaired and checked by reading, not compiled, not run on hardware
//
#include <hip/hip_runtime.h>

typedef __attribute__((ext_vector_type(16))) _Float16 v16h;
typedef __attribute__((ext_vector_type(8)))  float    v8f;

#define A_N  64
#define T_N  512
#define D_N  256
#define H_N  4
#define HD_N 64
#define W_ELEMS (D_N * D_N)                        // 65536
#define QKV_ELEMS ((size_t)A_N * H_N * T_N * HD_N) // 8388608

static __device__ __forceinline__ v8f wmma_f16(v16h a, v16h b, v8f c) {
  // D = A(16x32 f16) * B(32x16 f16) + C(16x16 f32)
  return __builtin_amdgcn_wmma_f32_16x16x32_f16(
      /*neg_a=*/false, a, /*neg_b=*/false, b,
      /*c_mod=*/(short)0, c, /*reuse_a=*/false, /*reuse_b=*/false);
}

// ---------------------------------------------------------------------------
// Kernel 1: convert weights f32 -> f16 (row-major W is already the B-matrix
// layout for Y = X @ W^T: B[k][n] = W[n][k] = Wh[n*256 + k]).
// ---------------------------------------------------------------------------
__global__ __launch_bounds__(256)
void cvt_weights(const float* __restrict__ Wq, const float* __restrict__ Wk,
                 const float* __restrict__ Wv, const float* __restrict__ Wo,
                 _Float16* __restrict__ Wh) {
  int i = blockIdx.x * 256 + threadIdx.x;        // grid = 256 blocks -> 65536
  Wh[i]               = (_Float16)Wq[i];
  Wh[W_ELEMS + i]     = (_Float16)Wk[i];
  Wh[2 * W_ELEMS + i] = (_Float16)Wv[i];
  Wh[3 * W_ELEMS + i] = (_Float16)Wo[i];
}

// ---------------------------------------------------------------------------
// Kernel 2: QKV projection. grid = (2048 m-tiles, 3 = q/k/v), block = 128.
// Wave w computes output columns [64w, 64w+64) == head w.
// Q,K stored [a,h,t,d] f16; V stored transposed [a,h,d,t] f16.
// K-loop fully unrolled so the scheduler can overlap loads with WMMAs.
// ---------------------------------------------------------------------------
__global__ __launch_bounds__(128)
void qkv_kernel(const float* __restrict__ X, const _Float16* __restrict__ Wh,
                const float* __restrict__ bq, const float* __restrict__ bk,
                const float* __restrict__ bv,
                _Float16* __restrict__ Qh, _Float16* __restrict__ Kh,
                _Float16* __restrict__ Vt) {
  const int wave = threadIdx.x >> 5;
  const int lane = threadIdx.x & 31;
  const int g = lane >> 4, n16 = lane & 15;
  const int m0 = blockIdx.x * 16;
  const int which = blockIdx.y;                  // 0=q 1=k 2=v
  const _Float16* W = Wh + (size_t)which * W_ELEMS;
  const float* bias = (which == 0) ? bq : (which == 1) ? bk : bv;
  const float* xrow = X + (size_t)(m0 + n16) * D_N;

  v8f acc[4] = {};
  #pragma unroll
  for (int k0 = 0; k0 < D_N; k0 += 32) {
    if (k0 + 32 < D_N)                           // prefetch next A chunk (WGP scope)
      __builtin_prefetch(xrow + k0 + 32 + 8 * g, 0, 3);
    // A-tile: X rows m0..m0+15, f32 -> f16, contiguous k = 16g + s (same order as B)
    v16h ax;
    const float* xr = xrow + k0;
    #pragma unroll
    for (int i = 0; i < 8; ++i) {
      ax[i]     = (_Float16)xr[16 * g + i];
      ax[8 + i] = (_Float16)xr[16 * g + 8 + i];
    }
    #pragma unroll
    for (int j = 0; j < 4; ++j) {
      v16h b = *(const v16h*)(W + (size_t)(wave * 64 + 16 * j + n16) * D_N +
                              k0 + 16 * g);
      acc[j] = wmma_f16(ax, b, acc[j]);
    }
  }
  float bsv[4];
  #pragma unroll
  for (int j = 0; j < 4; ++j) bsv[j] = bias[wave * 64 + 16 * j + n16];

  #pragma unroll
  for (int j = 0; j < 4; ++j) {
    #pragma unroll
    for (int r = 0; r < 8; ++r) {
      int mg = m0 + r + 8 * g;                   // global row (a*T + t)
      int aa = mg >> 9, t = mg & (T_N - 1);
      int d = 16 * j + n16;                      // head-local dim, h == wave
      _Float16 hv = (_Float16)(acc[j][r] + bsv[j]);
      size_t ah = (size_t)(aa * H_N + wave);
      if (which == 0)      Qh[(ah * T_N + t) * HD_N + d] = hv;
      else if (which == 1) Kh[(ah * T_N + t) * HD_N + d] = hv;
      else                 Vt[(ah * HD_N + d) * T_N + t] = hv;
    }
  }
}

// ---------------------------------------------------------------------------
// Kernel 3: flash-style masked attention. grid = (32, H, A), block = 32.
// ---------------------------------------------------------------------------
__global__ __launch_bounds__(32)
void attn_kernel(const _Float16* __restrict__ Qh, const _Float16* __restrict__ Kh,
                 const _Float16* __restrict__ Vt, const float* __restrict__ pos,
                 _Float16* __restrict__ AO,
                 const int* __restrict__ mdp, const int* __restrict__ twp) {
  __shared__ _Float16 Plds[16 * 32];             // one P tile, C->A relayout
  const int lane = threadIdx.x;
  const int g = lane >> 4, n16 = lane & 15;
  const int t0 = blockIdx.x * 16;
  const int h  = blockIdx.y;
  const int a  = blockIdx.z;
  const int   md  = *mdp;
  const float md2 = (float)(md * md);
  const int   tw  = *twp;
  const float NEG_INF = -__builtin_inff();
  const size_t base = (size_t)(a * H_N + h) * T_N * HD_N; // Q/K and Vt share size

  // Q A-tiles (K = 64 -> two 16x32 tiles), kept in registers all kernel
  v16h qa[2];
  {
    const _Float16* qrow = Qh + base + (size_t)(t0 + n16) * HD_N;
    #pragma unroll
    for (int kt = 0; kt < 2; ++kt) {
      const _Float16* p0 = qrow + 32 * kt + 16 * g;
      #pragma unroll
      for (int i = 0; i < 8; ++i) { qa[kt][i] = p0[i]; qa[kt][8 + i] = p0[8 + i]; }
    }
  }
  // query positions for this lane's rows t = t0 + 8g + r
  float qx[8], qy[8];
  #pragma unroll
  for (int r = 0; r < 8; ++r) {
    int t = t0 + 8 * g + r;
    qx[r] = pos[((size_t)a * T_N + t) * 2 + 0];
    qy[r] = pos[((size_t)a * T_N + t) * 2 + 1];
  }

  v8f O[4] = {};
  float mrow[8], lrow[8];
  #pragma unroll
  for (int r = 0; r < 8; ++r) { mrow[r] = NEG_INF; lrow[r] = 0.0f; }

  int slo = t0 - tw; if (slo < 0) slo = 0; slo &= ~31;
  int shi = t0 + 15 + tw; if (shi > T_N - 1) shi = T_N - 1;

  for (int s0 = slo; s0 <= shi; s0 += 32) {
    // prefetch next key chunk (K rows and V rows) into WGP cache
    if (s0 + 32 <= shi) {
      __builtin_prefetch(Kh + base + (size_t)(s0 + 32 + n16) * HD_N, 0, 3);
      __builtin_prefetch(Kh + base + (size_t)(s0 + 48 + n16) * HD_N, 0, 3);
      __builtin_prefetch(Vt + base + (size_t)(16 * (lane >> 3) + (lane & 7)) * T_N +
                             s0 + 32, 0, 3);
    }
    // ---- S = Q @ K^T (two 16x16 tiles over a 32-key chunk)
    v8f S[2];
    #pragma unroll
    for (int j = 0; j < 2; ++j) {
      v8f c = {};
      #pragma unroll
      for (int kt = 0; kt < 2; ++kt) {
        v16h b = *(const v16h*)(Kh + base + (size_t)(s0 + 16 * j + n16) * HD_N +
                                32 * kt + 16 * g);
        c = wmma_f16(qa[kt], b, c);
      }
      S[j] = c;
    }
    // ---- mask (spatial distance + time window), scale
    float sv[2][8];
    #pragma unroll
    for (int j = 0; j < 2; ++j) {
      int s = s0 + 16 * j + n16;
      float px = pos[((size_t)a * T_N + s) * 2 + 0];
      float py = pos[((size_t)a * T_N + s) * 2 + 1];
      #pragma unroll
      for (int r = 0; r < 8; ++r) {
        int t = t0 + 8 * g + r;
        float dx = qx[r] - px, dy = qy[r] - py;
        int dt = (t > s) ? (t - s) : (s - t);
        bool ok = (dx * dx + dy * dy <= md2) && (dt <= tw);
        sv[j][r] = ok ? S[j][r] * 0.125f : NEG_INF;   // 1/sqrt(64)
      }
    }
    // ---- online softmax (row = r+8g lives across a 16-lane half)
    float alpha[8], p[2][8];
    #pragma unroll
    for (int r = 0; r < 8; ++r) {
      float cm = fmaxf(sv[0][r], sv[1][r]);
      #pragma unroll
      for (int msk = 1; msk < 16; msk <<= 1)
        cm = fmaxf(cm, __shfl_xor(cm, msk, 32));
      float mnew = fmaxf(mrow[r], cm);
      alpha[r] = (mnew == NEG_INF) ? 1.0f : __expf(mrow[r] - mnew);
      p[0][r] = (sv[0][r] == NEG_INF) ? 0.0f : __expf(sv[0][r] - mnew);
      p[1][r] = (sv[1][r] == NEG_INF) ? 0.0f : __expf(sv[1][r] - mnew);
      float ps = p[0][r] + p[1][r];
      #pragma unroll
      for (int msk = 1; msk < 16; msk <<= 1)
        ps += __shfl_xor(ps, msk, 32);
      lrow[r] = lrow[r] * alpha[r] + ps;
      mrow[r] = mnew;
    }
    #pragma unroll
    for (int jn = 0; jn < 4; ++jn)
      #pragma unroll
      for (int r = 0; r < 8; ++r) O[jn][r] *= alpha[r];
    // ---- P: C-layout -> LDS -> A-layout (f16)
    #pragma unroll
    for (int j = 0; j < 2; ++j)
      #pragma unroll
      for (int r = 0; r < 8; ++r)
        Plds[(r + 8 * g) * 32 + 16 * j + n16] = (_Float16)p[j][r];
    v16h pa;
    {
      const _Float16* p0 = &Plds[n16 * 32 + 16 * g];
      #pragma unroll
      for (int i = 0; i < 8; ++i) { pa[i] = p0[i]; pa[8 + i] = p0[8 + i]; }
    }
    // ---- O += P @ V (V transposed in memory: contiguous along k=s)
    #pragma unroll
    for (int jn = 0; jn < 4; ++jn) {
      v16h vb = *(const v16h*)(Vt + base + (size_t)(16 * jn + n16) * T_N +
                               s0 + 16 * g);
      O[jn] = wmma_f16(pa, vb, O[jn]);
    }
  }

  // ---- finalize & store f16 [a,t,h*64+d]
  #pragma unroll
  for (int r = 0; r < 8; ++r) {
    float inv = 1.0f / lrow[r];                  // diagonal always allowed
    #pragma unroll
    for (int jn = 0; jn < 4; ++jn) O[jn][r] *= inv;
  }
  #pragma unroll
  for (int jn = 0; jn < 4; ++jn)
    #pragma unroll
    for (int r = 0; r < 8; ++r) {
      int t = t0 + 8 * g + r;
      AO[((size_t)a * T_N + t) * D_N + h * HD_N + 16 * jn + n16] =
          (_Float16)O[jn][r];
    }
}

// ---------------------------------------------------------------------------
// Kernel 4: output projection AO(f16) @ Wo^T + bo -> f32 d_out.
// ---------------------------------------------------------------------------
__global__ __launch_bounds__(128)
void oproj_kernel(const _Float16* __restrict__ AO, const _Float16* __restrict__ Wh,
                  const float* __restrict__ bo, float* __restrict__ out) {
  const int wave = threadIdx.x >> 5;
  const int lane = threadIdx.x & 31;
  const int g = lane >> 4, n16 = lane & 15;
  const int m0 = blockIdx.x * 16;
  const _Float16* W = Wh + (size_t)3 * W_ELEMS;
  const _Float16* arow = AO + (size_t)(m0 + n16) * D_N;

  v8f acc[4] = {};
  #pragma unroll
  for (int k0 = 0; k0 < D_N; k0 += 32) {
    if (k0 + 32 < D_N)                           // prefetch next A chunk (WGP scope)
      __builtin_prefetch(arow + k0 + 32 + 8 * g, 0, 3);
    v16h ax;
    const _Float16* ar = arow + k0;
    #pragma unroll
    for (int i = 0; i < 8; ++i) { ax[i] = ar[16 * g + i]; ax[8 + i] = ar[16 * g + 8 + i]; }
    #pragma unroll
    for (int j = 0; j < 4; ++j) {
      v16h b = *(const v16h*)(W + (size_t)(wave * 64 + 16 * j + n16) * D_N +
                              k0 + 16 * g);
      acc[j] = wmma_f16(ax, b, acc[j]);
    }
  }
  #pragma unroll
  for (int j = 0; j < 4; ++j) {
    float bv = bo[wave * 64 + 16 * j + n16];
    #pragma unroll
    for (int r = 0; r < 8; ++r) {
      int mg = m0 + r + 8 * g;
      out[(size_t)mg * D_N + wave * 64 + 16 * j + n16] = acc[j][r] + bv;
    }
  }
}

// ---------------------------------------------------------------------------
extern "C" void kernel_launch(void* const* d_in, const int* in_sizes, int n_in,
                              void* d_out, int out_size, void* d_ws, size_t ws_size,
                              hipStream_t stream) {
  (void)in_sizes; (void)n_in; (void)out_size; (void)ws_size;
  const float* X   = (const float*)d_in[0];
  const float* pos = (const float*)d_in[1];
  const float* Wq  = (const float*)d_in[2];
  const float* bq  = (const float*)d_in[3];
  const float* Wk  = (const float*)d_in[4];
  const float* bk  = (const float*)d_in[5];
  const float* Wv  = (const float*)d_in[6];
  const float* bv  = (const float*)d_in[7];
  const float* Wo  = (const float*)d_in[8];
  const float* bo  = (const float*)d_in[9];
  const int*   md  = (const int*)d_in[10];
  const int*   tw  = (const int*)d_in[11];
  float* out = (float*)d_out;

  // workspace layout (f16): 4 weights | Qh | Kh | Vt | AO  (~64.5 MiB total)
  _Float16* Wh = (_Float16*)d_ws;
  _Float16* Qh = Wh + (size_t)4 * W_ELEMS;
  _Float16* Kh = Qh + QKV_ELEMS;
  _Float16* Vt = Kh + QKV_ELEMS;
  _Float16* AO = Vt + QKV_ELEMS;

  cvt_weights<<<W_ELEMS / 256, 256, 0, stream>>>(Wq, Wk, Wv, Wo, Wh);
  qkv_kernel<<<dim3((A_N * T_N) / 16, 3), 128, 0, stream>>>(X, Wh, bq, bk, bv,
                                                            Qh, Kh, Vt);
  attn_kernel<<<dim3(T_N / 16, H_N, A_N), 32, 0, stream>>>(Qh, Kh, Vt, pos, AO,
                                                           md, tw);
  oproj_kernel<<<(A_N * T_N) / 16, 128, 0, stream>>>(AO, Wh, bo, out);
}
